// AffSkLayer_83313775607973
// MI455X (gfx1250) — matrix operand order repaired and checked by reading, not compiled, not run on hardware
//
#include <hip/hip_runtime.h>

// ---------------------------------------------------------------------------
// CDNA5 (gfx1250, wave32) WMMA types
// ---------------------------------------------------------------------------
typedef __attribute__((ext_vector_type(16))) __bf16 v16bf;
typedef __attribute__((ext_vector_type(8)))  float  v8f;

union FragBF {
    unsigned int u[8];
    uint4        q[2];
    v16bf        v;
};

__device__ __forceinline__ unsigned short f2bf(float f) {
    union { float f; unsigned int u; } v; v.f = f;
    unsigned int u = v.u;
    return (unsigned short)((u + 0x7FFFu + ((u >> 16) & 1u)) >> 16);
}
__device__ __forceinline__ float bf2f(unsigned short h) {
    union { unsigned int u; float f; } v; v.u = ((unsigned int)h) << 16;
    return v.f;
}

#define CC   256
#define RC   64
#define TT   8
#define NN   784          // 28*28
#define PP   6272         // TT*NN
#define BB   4
#define TOTE ((size_t)BB * CC * PP)   // 6,422,528
#define TOTR ((size_t)BB * RC * PP)   // 1,605,632
#define INV_TEMP (1.0f / 0.07f)

// ---------------------------------------------------------------------------
// f32 -> bf16 (weights)
// ---------------------------------------------------------------------------
__global__ void k_f2bf(const float* __restrict__ in, unsigned short* __restrict__ out, int n) {
    int i = blockIdx.x * blockDim.x + threadIdx.x;
    if (i < n) out[i] = f2bf(in[i]);
}

// ---------------------------------------------------------------------------
// Transpose input x [B][C][P] f32 -> Xh [B][P][C] bf16 (LDS 64x64 tile)
// ---------------------------------------------------------------------------
__global__ void k_transpose(const float* __restrict__ X, unsigned short* __restrict__ Xh)
{
    __shared__ unsigned short tile[64 * 72];
    const int p0 = blockIdx.x * 64;
    const int c0 = blockIdx.y * 64;
    const int b  = blockIdx.z;
    #pragma unroll
    for (int j = 0; j < 16; ++j) {
        int id = threadIdx.x + 256 * j;
        int c_l = id >> 6, p_l = id & 63;
        float v = X[(size_t)(b * CC + c0 + c_l) * PP + p0 + p_l];
        tile[c_l * 72 + p_l] = f2bf(v);
    }
    __syncthreads();
    #pragma unroll
    for (int j = 0; j < 16; ++j) {
        int id = threadIdx.x + 256 * j;
        int p_l = id >> 6, c_l = id & 63;
        Xh[((size_t)b * PP + p0 + p_l) * CC + c0 + c_l] = tile[c_l * 72 + p_l];
    }
}

// ---------------------------------------------------------------------------
// Position-major WMMA GEMM:  Y[b][p][o] = sum_c A[b][p][c] * W[o][c]
// A bf16 [B][P][C]; W bf16 row-major [O][C]; Y f32 [B][P][O].
// Both fragments are per-lane-contiguous -> direct b128 loads, no LDS.
// Block 128 = 4 waves x 16 positions; A fragments for full K kept in VGPRs.
// ---------------------------------------------------------------------------
template<int C>
__global__ void k_gemm(const unsigned short* __restrict__ Ah,
                       const unsigned short* __restrict__ Wm,
                       float* __restrict__ Yt,
                       int O, int relu)
{
    constexpr int KS = C / 32;
    const int lane = threadIdx.x & 31;
    const int wv   = threadIdx.x >> 5;
    const int half = lane >> 4;
    const int l15  = lane & 15;
    const int p0   = blockIdx.x * 64;
    const int b    = blockIdx.y;

    const int row = p0 + wv * 16 + l15;
    const unsigned short* Arow = Ah + ((size_t)b * PP + row) * C;

    FragBF afr[KS];
    #pragma unroll
    for (int ks = 0; ks < KS; ++ks) {
        afr[ks].q[0] = *reinterpret_cast<const uint4*>(Arow + 32 * ks + 8 * half);
        afr[ks].q[1] = *reinterpret_cast<const uint4*>(Arow + 32 * ks + 16 + 8 * half);
    }

    float* Yb = Yt + (size_t)b * PP * O;
    for (int ot = 0; ot < O; ot += 16) {
        const unsigned short* Wrow = Wm + (size_t)(ot + l15) * C;
        v8f acc = {};
        #pragma unroll
        for (int ks = 0; ks < KS; ++ks) {
            FragBF bfr;
            bfr.q[0] = *reinterpret_cast<const uint4*>(Wrow + 32 * ks + 8 * half);
            bfr.q[1] = *reinterpret_cast<const uint4*>(Wrow + 32 * ks + 16 + 8 * half);
            acc = __builtin_amdgcn_wmma_f32_16x16x32_bf16(
                false, afr[ks].v, false, bfr.v, (short)0, acc, false, false);
        }
        #pragma unroll
        for (int vg = 0; vg < 8; ++vg) {
            float v = acc[vg];
            if (relu) v = fmaxf(v, 0.0f);
            Yb[(size_t)(p0 + wv * 16 + vg + 8 * half) * O + ot + l15] = v;
        }
    }
}

// ---------------------------------------------------------------------------
// Channel L2 norm per position (contiguous), emit bf16 X1h and normalized NXh.
// One wave per position; lane handles 8 contiguous channels.
// ---------------------------------------------------------------------------
__global__ void k_normsplit(const float* __restrict__ X1,        // [B][P][C]
                            unsigned short* __restrict__ X1h,    // [B][P][C]
                            unsigned short* __restrict__ NXh)    // [B][P][C]
{
    int wave = (blockIdx.x * blockDim.x + threadIdx.x) >> 5;
    int lane = threadIdx.x & 31;
    if (wave >= BB * PP) return;
    const float* src = X1 + (size_t)wave * CC + lane * 8;
    float4 v0 = *reinterpret_cast<const float4*>(src);
    float4 v1 = *reinterpret_cast<const float4*>(src + 4);
    float ss = v0.x * v0.x + v0.y * v0.y + v0.z * v0.z + v0.w * v0.w
             + v1.x * v1.x + v1.y * v1.y + v1.z * v1.z + v1.w * v1.w;
    #pragma unroll
    for (int off = 16; off >= 1; off >>= 1) ss += __shfl_xor(ss, off, 32);
    float inv = 1.0f / fmaxf(sqrtf(ss), 1e-12f);

    union { unsigned short s[8]; uint4 q; } on, ox;
    float vv[8] = { v0.x, v0.y, v0.z, v0.w, v1.x, v1.y, v1.z, v1.w };
    #pragma unroll
    for (int j = 0; j < 8; ++j) { on.s[j] = f2bf(vv[j] * inv); ox.s[j] = f2bf(vv[j]); }
    *reinterpret_cast<uint4*>(NXh + (size_t)wave * CC + lane * 8) = on.q;
    *reinterpret_cast<uint4*>(X1h + (size_t)wave * CC + lane * 8) = ox.q;
}

// ---------------------------------------------------------------------------
// Gram + streaming sum-of-exp (fixed max = 1/TEMP, entries are cosine sims):
//   S[sh][b][t][n] = sum_m exp((G[n,m]-1)/TEMP),  G = nx_t^T nx_ts
// A fragments (full K=256) preloaded in VGPRs; B tiles (16 positions x 256c,
// 8KB) double-buffered in LDS via async global->LDS loads (ASYNCcnt).
// Cross-lane reduction hoisted out of the m-loop: each lane accumulates the
// exp-sums of its own columns, one 16-lane butterfly at the end.
// ---------------------------------------------------------------------------
__global__ void k_gram(const unsigned short* __restrict__ NXh,  // [B][P][C]
                       float* __restrict__ S)                   // [3][32][NN]
{
    __shared__ unsigned short BT[2][16 * 256];   // 2 x 8KB

    const int lane = threadIdx.x & 31;
    const int wv   = threadIdx.x >> 5;
    const int half = lane >> 4;
    const int l15  = lane & 15;
    const int blk  = blockIdx.x;
    const int bt   = blockIdx.y;
    const int sh   = blockIdx.z;                 // 0=prev, 1=next, 2=same
    const int b = bt >> 3, t = bt & 7;
    int ts = t + (sh == 0 ? -1 : (sh == 1 ? 1 : 0));
    ts = min(TT - 1, max(0, ts));

    int rowc = min(blk * 64 + wv * 16 + l15, NN - 1);
    const unsigned short* Arow = NXh + ((size_t)b * PP + t * NN + rowc) * CC;
    FragBF afr[8];
    #pragma unroll
    for (int ks = 0; ks < 8; ++ks) {
        afr[ks].q[0] = *reinterpret_cast<const uint4*>(Arow + 32 * ks + 8 * half);
        afr[ks].q[1] = *reinterpret_cast<const uint4*>(Arow + 32 * ks + 16 + 8 * half);
    }

    const unsigned short* Bbase = NXh + ((size_t)b * PP + ts * NN) * CC;
    const unsigned lds0 = (unsigned)(size_t)(&BT[0][0]);   // flat->LDS offset (low 32b)

    // Async-stage tile 0 into buffer 0: 512 chunks of 16B, 4 per thread.
    #pragma unroll
    for (int r = 0; r < 4; ++r) {
        int qq = threadIdx.x + 128 * r;
        unsigned goff = (unsigned)((((qq >> 5)) * CC + (qq & 31) * 8) * 2);
        unsigned loff = lds0 + (unsigned)(qq * 16);
        asm volatile("global_load_async_to_lds_b128 %0, %1, %2"
                     :: "v"(loff), "v"(goff), "s"(Bbase) : "memory");
    }

    float sacc[8];
    #pragma unroll
    for (int i = 0; i < 8; ++i) sacc[i] = 0.f;

    for (int mt = 0; mt < 49; ++mt) {
        if (mt + 1 < 49) {
            // Prefetch next tile into the other buffer, then wait for current.
            int mbase = (mt + 1) * 16;
            unsigned bufo = ((mt + 1) & 1) ? (unsigned)(16 * 256 * 2) : 0u;
            #pragma unroll
            for (int r = 0; r < 4; ++r) {
                int qq = threadIdx.x + 128 * r;
                unsigned goff = (unsigned)(((mbase + (qq >> 5)) * CC + (qq & 31) * 8) * 2);
                unsigned loff = lds0 + bufo + (unsigned)(qq * 16);
                asm volatile("global_load_async_to_lds_b128 %0, %1, %2"
                             :: "v"(loff), "v"(goff), "s"(Bbase) : "memory");
            }
            asm volatile("s_wait_asynccnt 0x4" ::: "memory");
        } else {
            asm volatile("s_wait_asynccnt 0x0" ::: "memory");
        }
        __syncthreads();

        const unsigned short* Brow = &BT[mt & 1][l15 * 256];
        v8f acc = {};
        #pragma unroll
        for (int ks = 0; ks < 8; ++ks) {
            FragBF bfr;
            bfr.q[0] = *reinterpret_cast<const uint4*>(Brow + 32 * ks + 8 * half);
            bfr.q[1] = *reinterpret_cast<const uint4*>(Brow + 32 * ks + 16 + 8 * half);
            acc = __builtin_amdgcn_wmma_f32_16x16x32_bf16(
                false, afr[ks].v, false, bfr.v, (short)0, acc, false, false);
        }

        // Lane-private accumulation only; reduction deferred to after the loop.
        #pragma unroll
        for (int vg = 0; vg < 8; ++vg)
            sacc[vg] += __expf((acc[vg] - 1.0f) * INV_TEMP);

        __syncthreads();   // buffer consumed; safe to overwrite next iteration
    }

    // One butterfly reduction over the 16 column-lanes per half.
    #pragma unroll
    for (int vg = 0; vg < 8; ++vg) {
        float e = sacc[vg];
        e += __shfl_xor(e, 1, 32);
        e += __shfl_xor(e, 2, 32);
        e += __shfl_xor(e, 4, 32);
        e += __shfl_xor(e, 8, 32);
        sacc[vg] = e;
    }

    if (l15 == 0) {
        int rbase = blk * 64 + wv * 16 + 8 * half;
        float* Sout = S + ((size_t)sh * 32 + bt) * NN;
        #pragma unroll
        for (int vg = 0; vg < 8; ++vg) {
            int r = rbase + vg;
            if (r < NN) Sout[r] = sacc[vg];
        }
    }
}

// ---------------------------------------------------------------------------
// Softmax-diagonal weight: w = exp((diag-1)/T) / S; diag over contiguous c.
// ---------------------------------------------------------------------------
__global__ void k_weight(const unsigned short* __restrict__ NXh,
                         const float* __restrict__ S,
                         float* __restrict__ Wgt)
{
    int i = blockIdx.x * blockDim.x + threadIdx.x;
    if (i >= 3 * 32 * NN) return;
    int n = i % NN;
    int bt = (i / NN) % 32;
    int sh = i / (NN * 32);
    int b = bt >> 3, t = bt & 7;
    int ts = t + (sh == 0 ? -1 : (sh == 1 ? 1 : 0));
    ts = min(TT - 1, max(0, ts));

    const unsigned short* a  = NXh + ((size_t)b * PP + t  * NN + n) * CC;
    const unsigned short* c2 = NXh + ((size_t)b * PP + ts * NN + n) * CC;
    float d = 0.f;
    for (int c = 0; c < CC; c += 2) {
        unsigned int ua = *reinterpret_cast<const unsigned int*>(a + c);
        unsigned int uc = *reinterpret_cast<const unsigned int*>(c2 + c);
        d += bf2f((unsigned short)(ua & 0xffffu)) * bf2f((unsigned short)(uc & 0xffffu));
        d += bf2f((unsigned short)(ua >> 16))     * bf2f((unsigned short)(uc >> 16));
    }
    Wgt[i] = __expf((d - 1.0f) * INV_TEMP) / S[i];
}

// ---------------------------------------------------------------------------
// feats[b][p][c] = (1/3) * sum_s w_s[bt,n] * X1[b][p_shift][c]
// ---------------------------------------------------------------------------
__global__ void k_feats(const float* __restrict__ X1,   // [B][P][C]
                        const float* __restrict__ Wgt,  // [3][32][NN]
                        float* __restrict__ FE)         // [B][P][C]
{
    size_t i = (size_t)blockIdx.x * blockDim.x + threadIdx.x;
    if (i >= TOTE) return;
    int c = (int)(i % CC);
    int p = (int)((i / CC) % PP);
    int b = (int)(i / ((size_t)CC * PP));
    int n = p % NN, t = p / NN;
    int bt = b * TT + t;
    int tp = max(0, t - 1), tn = min(TT - 1, t + 1);
    const float* xb = X1 + (size_t)b * PP * CC;
    float acc = Wgt[(0 * 32 + bt) * NN + n] * xb[((size_t)tp * NN + n) * CC + c]
              + Wgt[(1 * 32 + bt) * NN + n] * xb[((size_t)tn * NN + n) * CC + c]
              + Wgt[(2 * 32 + bt) * NN + n] * xb[((size_t)t  * NN + n) * CC + c];
    FE[i] = acc * (1.0f / 3.0f);
}

// ---------------------------------------------------------------------------
// Depthwise 9x3x3, dilations 1/2/3 averaged; channel-contiguous reads.
// ---------------------------------------------------------------------------
__global__ void k_dwconv(const float* __restrict__ XXt,  // [B][P][RC]
                         const float* __restrict__ W1,
                         const float* __restrict__ W2,
                         const float* __restrict__ W3,
                         unsigned short* __restrict__ AGGh) // [B][P][RC]
{
    size_t i = (size_t)blockIdx.x * blockDim.x + threadIdx.x;
    if (i >= TOTR) return;
    int ch  = (int)(i % RC);
    int pos = (int)(i / RC);
    int xw = pos % 28;
    int yh = (pos / 28) % 28;
    int t  = (pos / NN) % TT;
    int b  = pos / PP;

    const float* xb = XXt + (size_t)b * PP * RC + ch;
    float acc = 0.f;
    #pragma unroll
    for (int d = 1; d <= 3; ++d) {
        const float* Wp = (d == 1 ? W1 : (d == 2 ? W2 : W3)) + ch * 81;
        for (int kt = 0; kt < 9; ++kt) {
            int tt2 = t + kt - 4;
            if (tt2 < 0 || tt2 >= TT) continue;
            const float* plane = xb + (size_t)tt2 * NN * RC;
            #pragma unroll
            for (int ky = 0; ky < 3; ++ky) {
                int yy = yh + (ky - 1) * d;
                if ((unsigned)yy >= 28u) continue;
                #pragma unroll
                for (int kx = 0; kx < 3; ++kx) {
                    int xx2 = xw + (kx - 1) * d;
                    if ((unsigned)xx2 >= 28u) continue;
                    acc += plane[(size_t)(yy * 28 + xx2) * RC] * Wp[kt * 9 + ky * 3 + kx];
                }
            }
        }
    }
    AGGh[i] = f2bf(acc * (1.0f / 3.0f));
}

// ---------------------------------------------------------------------------
// FSh = bf16(feats * score) (same position-major layout)
// ---------------------------------------------------------------------------
__global__ void k_mulbf(const float* __restrict__ FE,
                        const float* __restrict__ SC,
                        unsigned short* __restrict__ FSh)
{
    size_t i = (size_t)blockIdx.x * blockDim.x + threadIdx.x;
    if (i < TOTE) FSh[i] = f2bf(FE[i] * SC[i]);
}

// ---------------------------------------------------------------------------
// BN statistics: block per channel; y = x (channel-major) + F (position-major)
// ---------------------------------------------------------------------------
__global__ void k_bnstats(const float* __restrict__ X,   // [B][C][P]
                          const float* __restrict__ F,   // [B][P][C]
                          float* __restrict__ stats)
{
    int c = blockIdx.x;
    __shared__ float s1[256], s2[256];
    float a = 0.f, q = 0.f;
    for (int idx = threadIdx.x; idx < BB * PP; idx += 256) {
        int b = idx / PP, p = idx % PP;
        float y = X[(size_t)(b * CC + c) * PP + p] + F[((size_t)b * PP + p) * CC + c];
        a += y; q += y * y;
    }
    s1[threadIdx.x] = a; s2[threadIdx.x] = q;
    __syncthreads();
    for (int st = 128; st >= 1; st >>= 1) {
        if (threadIdx.x < st) {
            s1[threadIdx.x] += s1[threadIdx.x + st];
            s2[threadIdx.x] += s2[threadIdx.x + st];
        }
        __syncthreads();
    }
    if (threadIdx.x == 0) {
        float mean = s1[0] / (float)(BB * PP);
        float var  = s2[0] / (float)(BB * PP) - mean * mean;
        stats[c] = mean;
        stats[CC + c] = var;
    }
}

__global__ void k_bnapply(const float* __restrict__ X,   // [B][C][P]
                          const float* __restrict__ F,   // [B][P][C]
                          const float* __restrict__ stats,
                          const float* __restrict__ gamma,
                          const float* __restrict__ beta,
                          float* __restrict__ out)       // [B][C][P]
{
    size_t i = (size_t)blockIdx.x * blockDim.x + threadIdx.x;
    if (i >= TOTE) return;
    int p = (int)(i % PP);
    int c = (int)((i / PP) % CC);
    int b = (int)(i / ((size_t)CC * PP));
    float mean = stats[c], var = stats[CC + c];
    float y = X[i] + F[((size_t)b * PP + p) * CC + c];
    float v = gamma[c] * (y - mean) * rsqrtf(var + 1e-5f) + beta[c];
    out[i] = fmaxf(v, 0.0f);
}

// ---------------------------------------------------------------------------
// Launch pipeline (all position-major intermediates)
// ---------------------------------------------------------------------------
extern "C" void kernel_launch(void* const* d_in, const int* in_sizes, int n_in,
                              void* d_out, int out_size, void* d_ws, size_t ws_size,
                              hipStream_t stream)
{
    (void)in_sizes; (void)n_in; (void)out_size; (void)ws_size;
    const float* x     = (const float*)d_in[0];
    const float* Wdc2  = (const float*)d_in[1];
    const float* Wup2  = (const float*)d_in[2];
    const float* Wdc   = (const float*)d_in[3];
    const float* Wsa1  = (const float*)d_in[4];
    const float* Wsa2  = (const float*)d_in[5];
    const float* Wsa3  = (const float*)d_in[6];
    const float* Wback = (const float*)d_in[7];
    const float* gamma = (const float*)d_in[8];
    const float* beta  = (const float*)d_in[9];
    float* out = (float*)d_out;

    char* ws = (char*)d_ws;
    size_t off = 0;
    auto take = [&](size_t bytes) -> char* {
        char* r = ws + off;
        off = (off + bytes + 255) & ~(size_t)255;
        return r;
    };

    float*          X1    = (float*)take(TOTE * 4);          // [B][P][C]
    unsigned short* Xh    = (unsigned short*)take(TOTE * 2); // [B][P][C]
    unsigned short* X1h   = (unsigned short*)take(TOTE * 2);
    unsigned short* NXh   = (unsigned short*)take(TOTE * 2);
    float*          FE    = (float*)take(TOTE * 4);
    float*          XXt   = (float*)take(TOTR * 4);          // [B][P][RC]
    unsigned short* AGGh  = (unsigned short*)take(TOTR * 2);
    float*          SCt   = (float*)take(TOTE * 4);
    float*          S     = (float*)take((size_t)3 * 32 * NN * 4);
    float*          Wgt   = (float*)take((size_t)3 * 32 * NN * 4);
    unsigned short* Wdc2h = (unsigned short*)take(65536 * 2);
    unsigned short* Wup2h = (unsigned short*)take(65536 * 2);
    unsigned short* Wdch  = (unsigned short*)take(16384 * 2);
    unsigned short* Wbackh= (unsigned short*)take(16384 * 2);
    float*          stats = (float*)take(512 * 4);
    float*          Ft    = X1;    // reuse: X1 last read in k_feats
    unsigned short* FSh   = X1h;   // reuse: X1h last read in xx-GEMM

    // weight conversions (row-major [O][C] kept; used directly as WMMA B)
    k_f2bf<<<256, 256, 0, stream>>>(Wdc2, Wdc2h, 65536);
    k_f2bf<<<256, 256, 0, stream>>>(Wup2, Wup2h, 65536);
    k_f2bf<<<64, 256, 0, stream>>>(Wdc, Wdch, 16384);
    k_f2bf<<<64, 256, 0, stream>>>(Wback, Wbackh, 16384);

    // x -> position-major bf16
    k_transpose<<<dim3(98, 4, 4), 256, 0, stream>>>(x, Xh);
    // X1 = x @ W_dc2^T
    k_gemm<256><<<dim3(98, BB), 128, 0, stream>>>(Xh, Wdc2h, X1, 256, 0);
    // norms + bf16 copies
    k_normsplit<<<3136, 256, 0, stream>>>(X1, X1h, NXh);
    // Gram sum-of-exp (dominant WMMA work, async double-buffered B tiles)
    k_gram<<<dim3(13, 32, 3), 128, 0, stream>>>(NXh, S);
    // diag softmax weights
    k_weight<<<(3 * 32 * NN + 255) / 256, 256, 0, stream>>>(NXh, S, Wgt);
    // feats
    k_feats<<<(unsigned)((TOTE + 255) / 256), 256, 0, stream>>>(X1, Wgt, FE);
    // xx = X1 @ W_dc^T
    k_gemm<256><<<dim3(98, BB), 128, 0, stream>>>(X1h, Wdch, XXt, 64, 0);
    // dilated depthwise convs -> agg (bf16)
    k_dwconv<<<(unsigned)((TOTR + 255) / 256), 256, 0, stream>>>(XXt, Wsa1, Wsa2, Wsa3, AGGh);
    // score = relu(agg @ W_back^T)
    k_gemm<64><<<dim3(98, BB), 128, 0, stream>>>(AGGh, Wbackh, SCt, 256, 1);
    // FSh = bf16(feats * score)
    k_mulbf<<<(unsigned)((TOTE + 255) / 256), 256, 0, stream>>>(FE, SCt, FSh);
    // F = FS @ W_up2^T
    k_gemm<256><<<dim3(98, BB), 128, 0, stream>>>(FSh, Wup2h, Ft, 256, 0);
    // BN + relu
    k_bnstats<<<256, 256, 0, stream>>>(x, Ft, stats);
    k_bnapply<<<(unsigned)((TOTE + 255) / 256), 256, 0, stream>>>(x, Ft, stats, gamma, beta, out);
}
